// SparseExpertRouter_21182778703905
// MI455X (gfx1250) — matrix operand order repaired and checked by reading, not compiled, not run on hardware
//
#include <hip/hip_runtime.h>

#define TOKENS 16384
#define HID    4096
#define PH     256
#define NE     64
#define NC     16
#define NTGT   24
#define BM     64

typedef __attribute__((ext_vector_type(16))) __bf16 v16bf;
typedef __attribute__((ext_vector_type(8)))  __bf16 v8bf;
typedef __attribute__((ext_vector_type(8)))  float  v8f;

#define LDA_S 40    // A tile row stride (bf16 elems): 32 + 8 pad, 80B rows (16B aligned chunks)
#define LDB_S 40    // B tile row stride
#define LDH_S 264   // h row stride: 256 + 8 pad, 528B rows (16B aligned chunks)

__device__ __forceinline__ v16bf cat16(v8bf a, v8bf b) {
  return __builtin_shufflevector(a, b, 0,1,2,3,4,5,6,7,8,9,10,11,12,13,14,15);
}

__device__ __forceinline__ v8bf cvt8(float4 a, float4 b) {
  v8bf r;
  r[0]=(__bf16)a.x; r[1]=(__bf16)a.y; r[2]=(__bf16)a.z; r[3]=(__bf16)a.w;
  r[4]=(__bf16)b.x; r[5]=(__bf16)b.y; r[6]=(__bf16)b.z; r[7]=(__bf16)b.w;
  return r;
}

__global__ __launch_bounds__(256)
void router_fused_kernel(const float* __restrict__ x,
                         const float* __restrict__ w1,
                         const float* __restrict__ b1,
                         const float* __restrict__ w2,
                         const float* __restrict__ b2,
                         const float* __restrict__ ebias,
                         const float* __restrict__ cooc,
                         float* __restrict__ out)
{
  __shared__ __align__(16) __bf16 lH[BM * LDH_S];           // 33792 B, h (bf16)
  __shared__ __align__(16) unsigned char smemB[25600];      // stage1 A+B tiles, later logits
  __bf16* lA  = (__bf16*)smemB;                             // 64 x LDA_S bf16 (5120 B)
  __bf16* lB  = (__bf16*)(smemB + BM * LDA_S * 2);          // 256 x LDB_S bf16 (20480 B)
  float*  lLg = (float*)smemB;                              // 64 x 64 f32 logits (16384 B)

  const int t  = threadIdx.x;
  const int w  = t >> 5;          // wave 0..7
  const int l  = t & 31;          // lane
  const int lo = l & 15;
  const int kh = l >> 4;          // K-half select per ISA fragment layout
  const int m0 = blockIdx.x * BM;

  // ---------------- Stage 1: h = relu(x @ w1^T + b1), bf16 WMMA ----------------
  const int wm = w & 3;           // wave row tile: rows wm*16 .. +15
  const int wn = w >> 2;          // wave col half: cols wn*128 .. +127

  const v8f vzero = {0.f,0.f,0.f,0.f,0.f,0.f,0.f,0.f};
  v8f acc[8];
  #pragma unroll
  for (int j = 0; j < 8; ++j) acc[j] = vzero;

  const int arow = t >> 2;        // 0..63
  const int acg  = t & 3;         // 0..3 : 8-float chunk of the 32-wide K tile
  const float* xrow  = x  + (size_t)(m0 + arow) * HID + acg * 8;
  const float* w1row = w1 + (size_t)t * HID;

  // --- software pipeline: preload tile kk=0 into registers ---
  float4 fA0, fA1, fB[8];
  {
    const float4* pa = (const float4*)xrow;
    fA0 = pa[0]; fA1 = pa[1];
    const float4* pb = (const float4*)w1row;
    #pragma unroll
    for (int qd = 0; qd < 8; ++qd) fB[qd] = pb[qd];
  }

  for (int kk = 0; kk < HID; kk += 32) {
    // convert & stage current tile (s_wait_loadcnt lands here, one iter after issue)
    *(v8bf*)&lA[arow * LDA_S + acg * 8] = cvt8(fA0, fA1);
    #pragma unroll
    for (int qd = 0; qd < 4; ++qd)
      *(v8bf*)&lB[t * LDB_S + qd * 8] = cvt8(fB[2*qd], fB[2*qd+1]);
    __syncthreads();

    // issue NEXT tile's global loads now; they fly under the WMMAs below
    if (kk + 32 < HID) {
      const float4* pa = (const float4*)(xrow + kk + 32);
      fA0 = pa[0]; fA1 = pa[1];
      const float4* pb = (const float4*)(w1row + kk + 32);
      #pragma unroll
      for (int qd = 0; qd < 8; ++qd) fB[qd] = pb[qd];
    }

    // A fragment: lane(lo,kh) holds K = {kh*8+0..7, 16+kh*8+0..7} of row wm*16+lo
    v8bf a0 = *(const v8bf*)&lA[(wm*16 + lo) * LDA_S + kh * 8];
    v8bf a1 = *(const v8bf*)&lA[(wm*16 + lo) * LDA_S + 16 + kh * 8];
    v16bf af = cat16(a0, a1);

    #pragma unroll
    for (int j = 0; j < 8; ++j) {
      int n = wn * 128 + j * 16 + lo;
      // B fragment: lane holds K = {kh*8+0..7, 16+kh*8+0..7} of column n
      v8bf b0 = *(const v8bf*)&lB[n * LDB_S + kh * 8];
      v8bf b1v = *(const v8bf*)&lB[n * LDB_S + 16 + kh * 8];
      v16bf bf = cat16(b0, b1v);
      acc[j] = __builtin_amdgcn_wmma_f32_16x16x32_bf16(
          false, af, false, bf, (short)0, acc[j], false, false);
    }
    __syncthreads();
  }

  // epilogue: +b1, relu, store bf16 h to LDS.  C layout: elem v -> row = v + 8*kh
  #pragma unroll
  for (int j = 0; j < 8; ++j) {
    int n = wn * 128 + j * 16 + lo;
    float bias = b1[n];
    #pragma unroll
    for (int v = 0; v < 8; ++v) {
      int row = wm * 16 + kh * 8 + v;
      float h = acc[j][v] + bias;
      lH[row * LDH_S + n] = (__bf16)(h > 0.f ? h : 0.f);
    }
  }
  __syncthreads();

  // ---------------- Stage 2: logits = h @ w2^T + b2 + expert_bias ----------------
  const int m2 = (w & 3) * 16;    // 4 row tiles
  const int n2 = (w >> 2) * 32;   // 2 col pairs
  v8f acc2[2];
  acc2[0] = vzero; acc2[1] = vzero;

  #pragma unroll
  for (int k2 = 0; k2 < PH; k2 += 32) {
    int r = m2 + lo;
    v8bf a0 = *(const v8bf*)&lH[r * LDH_S + k2 + kh * 16];
    v8bf a1 = *(const v8bf*)&lH[r * LDH_S + k2 + kh * 16 + 8];
    v16bf af = cat16(a0, a1);
    #pragma unroll
    for (int jj = 0; jj < 2; ++jj) {
      int n = n2 + jj * 16 + lo;
      const float4* wp = (const float4*)(w2 + n * PH + k2 + kh * 16);
      float4 q0 = wp[0], q1 = wp[1], q2 = wp[2], q3 = wp[3];
      v16bf bf;
      bf[0]=(__bf16)q0.x;  bf[1]=(__bf16)q0.y;  bf[2]=(__bf16)q0.z;  bf[3]=(__bf16)q0.w;
      bf[4]=(__bf16)q1.x;  bf[5]=(__bf16)q1.y;  bf[6]=(__bf16)q1.z;  bf[7]=(__bf16)q1.w;
      bf[8]=(__bf16)q2.x;  bf[9]=(__bf16)q2.y;  bf[10]=(__bf16)q2.z; bf[11]=(__bf16)q2.w;
      bf[12]=(__bf16)q3.x; bf[13]=(__bf16)q3.y; bf[14]=(__bf16)q3.z; bf[15]=(__bf16)q3.w;
      acc2[jj] = __builtin_amdgcn_wmma_f32_16x16x32_bf16(
          false, af, false, bf, (short)0, acc2[jj], false, false);
    }
  }

  #pragma unroll
  for (int jj = 0; jj < 2; ++jj) {
    int n = n2 + jj * 16 + lo;
    float bias = b2[n] + ebias[n];           // TEMP == 1.0
    #pragma unroll
    for (int v = 0; v < 8; ++v) {
      int row = m2 + kh * 8 + v;
      lLg[row * NE + n] = acc2[jj][v] + bias;
    }
  }
  __syncthreads();

  // ---------------- Stage 3: top-16, cooc expand, top-8, emit ----------------
  const float NEG_INF = -__builtin_inff();
  for (int q = 0; q < 8; ++q) {
    int tok = w * 8 + q;
    float L0 = lLg[tok * NE + l];            // expert l
    float L1 = lLg[tok * NE + 32 + l];       // expert l+32
    bool sel0 = false, sel1 = false;
    float cs[NC]; int ci[NC];
    float out_sc = 0.f; int out_id = 0;

    #pragma unroll
    for (int it = 0; it < NC; ++it) {
      float a0 = sel0 ? NEG_INF : L0;
      float a1 = sel1 ? NEG_INF : L1;
      float v; int i;
      if (a0 >= a1) { v = a0; i = l; } else { v = a1; i = l + 32; }
      #pragma unroll
      for (int off = 16; off > 0; off >>= 1) {
        float ov = __shfl_xor(v, off, 32);
        int   oi = __shfl_xor(i, off, 32);
        if (ov > v || (ov == v && oi < i)) { v = ov; i = oi; }
      }
      cs[it] = v; ci[it] = i;
      if (i == l)      sel0 = true;
      if (i == l + 32) sel1 = true;
      if (l == it) { out_sc = v; out_id = i; }
    }

    // cooc_scores[e] = sum_i cs[i] * cooc[ci[i]][e]   (cooc is 16 KB -> L2 resident)
    float c0 = 0.f, c1 = 0.f;
    #pragma unroll
    for (int it = 0; it < NC; ++it) {
      const float* row = cooc + ci[it] * NE;
      c0 += cs[it] * row[l];
      c1 += cs[it] * row[l + 32];
    }
    if (sel0) c0 = NEG_INF;
    if (sel1) c1 = NEG_INF;

    #pragma unroll
    for (int it = 0; it < (NTGT - NC); ++it) {
      float v; int i;
      if (c0 >= c1) { v = c0; i = l; } else { v = c1; i = l + 32; }
      #pragma unroll
      for (int off = 16; off > 0; off >>= 1) {
        float ov = __shfl_xor(v, off, 32);
        int   oi = __shfl_xor(i, off, 32);
        if (ov > v || (ov == v && oi < i)) { v = ov; i = oi; }
      }
      if (i == l)      c0 = NEG_INF;
      if (i == l + 32) c1 = NEG_INF;
      if (l == NC + it) { out_sc = v; out_id = i; }
    }

    int gt = m0 + tok;
    if (l < NTGT) {
      out[(size_t)gt * NTGT + l] = (float)out_id;                         // expanded_ids
      out[(size_t)TOKENS * NTGT + (size_t)gt * NTGT + l] = out_sc;        // expanded_scores
    }
  }
}

extern "C" void kernel_launch(void* const* d_in, const int* in_sizes, int n_in,
                              void* d_out, int out_size, void* d_ws, size_t ws_size,
                              hipStream_t stream) {
  (void)in_sizes; (void)n_in; (void)out_size; (void)d_ws; (void)ws_size;
  const float* x     = (const float*)d_in[0];
  const float* w1    = (const float*)d_in[1];
  const float* b1    = (const float*)d_in[2];
  const float* w2    = (const float*)d_in[3];
  const float* b2    = (const float*)d_in[4];
  const float* ebias = (const float*)d_in[5];
  const float* cooc  = (const float*)d_in[6];
  router_fused_kernel<<<dim3(TOKENS / BM), dim3(256), 0, stream>>>(
      x, w1, b1, w2, b2, ebias, cooc, (float*)d_out);
}
